// GraphAutoencoder_8839042695242
// MI455X (gfx1250) — matrix operand order repaired and checked by reading, not compiled, not run on hardware
//
#include <hip/hip_runtime.h>
#include <hip/hip_bf16.h>

// MI455X / gfx1250, wave32. Matrix ops via V_WMMA_F32_16X16X32_BF16.

typedef __attribute__((ext_vector_type(16))) __bf16 v16bf;
typedef __attribute__((ext_vector_type(8)))  __bf16 v8bf;
typedef __attribute__((ext_vector_type(8)))  float  v8f;

#define IN_DIM 512
#define HID    256
#define LAT    128
#define KPROP  10
#define ALPHA  0.1f
#define BN_EPS 1e-5f

static __device__ __forceinline__ __bf16 f2bf(float f) {
  // round-to-nearest-even fp32 -> bf16
  unsigned u = __builtin_bit_cast(unsigned, f);
  unsigned short r = (unsigned short)((u + 0x7FFFu + ((u >> 16) & 1u)) >> 16);
  return __builtin_bit_cast(__bf16, r);
}

// ---------------- degree / norm ----------------
__global__ void k_fill(float* __restrict__ p, float v, size_t cnt) {
  size_t i = (size_t)blockIdx.x * blockDim.x + threadIdx.x;
  if (i < cnt) p[i] = v;
}

__global__ void k_deg_edges(const int* __restrict__ dst, float* __restrict__ deg, int e) {
  int i = blockIdx.x * blockDim.x + threadIdx.x;
  if (i < e) atomicAdd(&deg[dst[i]], 1.0f);
}

__global__ void k_dinv(float* __restrict__ d, int n) {
  int i = blockIdx.x * blockDim.x + threadIdx.x;
  if (i < n) d[i] = rsqrtf(fmaxf(d[i], 1.0f));
}

__global__ void k_enorm(const int* __restrict__ src, const int* __restrict__ dst,
                        const float* __restrict__ dinv, float* __restrict__ enorm, int e) {
  int i = blockIdx.x * blockDim.x + threadIdx.x;
  if (i < e) enorm[i] = dinv[src[i]] * dinv[dst[i]];
}

// ---------------- one-shot precision conversion / weight transpose ----------------
__global__ void k_f32_to_bf16(const float* __restrict__ in, __bf16* __restrict__ out, size_t cnt) {
  size_t i = (size_t)blockIdx.x * blockDim.x + threadIdx.x;
  if (i < cnt) out[i] = f2bf(in[i]);
}

// Bt[nc*Kd + k] = bf16(B[k*Nc + nc])  (weights are tiny; done once per GEMM)
__global__ void k_transpose_bf16(const float* __restrict__ B, __bf16* __restrict__ Bt,
                                 int Kd, int Nc) {
  int i = blockIdx.x * blockDim.x + threadIdx.x;
  if (i < Kd * Nc) {
    int nc = i / Kd, k = i % Kd;
    Bt[i] = f2bf(B[(size_t)k * Nc + nc]);
  }
}

// ---------------- bf16 WMMA GEMM: C[M,Nc] = A[M,Kd] @ B[Kd,Nc] ----------------
// A row-major bf16, Bt = B^T row-major bf16 (K contiguous). One wave computes a
// 16x64 output strip: 4 WMMA tiles per K-step sharing the A fragment.
// Requires M % 16 == 0 (50000 = 3125*16), Nc % 64 == 0, Kd % 32 == 0.
__global__ __launch_bounds__(256) void k_gemm_bf16(
    const __bf16* __restrict__ A, const __bf16* __restrict__ Bt, float* __restrict__ C,
    int M, int Kd, int Nc) {
  int gw   = (int)((blockIdx.x * blockDim.x + threadIdx.x) >> 5);  // global wave id
  int lane = (int)(threadIdx.x & 31);
  int ntn = Nc >> 6;                 // 64-column strips
  int mt  = M >> 4;
  if (gw >= mt * ntn) return;        // wave-uniform: EXEC stays all-1 for WMMA
  int tm = (gw / ntn) << 4;
  int tn = (gw % ntn) << 6;
  int l15 = lane & 15;
  int hi  = lane >> 4;

  const __bf16* ap = A + (size_t)(tm + l15) * Kd + (hi << 3);   // K = hi*8 + j / +16
  const __bf16* bp0 = Bt + (size_t)(tn + l15) * Kd + (hi << 3); // K = hi*8 + j / +16
  const size_t bstride = (size_t)16 * Kd;                        // next 16-col tile

  v8f acc0 = {}, acc1 = {}, acc2 = {}, acc3 = {};
  for (int kb = 0; kb < Kd; kb += 32) {
    __builtin_prefetch(ap + kb + 32, 0, 1);   // global_prefetch_b8 on the A stream
    v8bf a0 = *(const v8bf*)(ap + kb);
    v8bf a1 = *(const v8bf*)(ap + kb + 16);
    v16bf a = __builtin_shufflevector(a0, a1, 0,1,2,3,4,5,6,7,8,9,10,11,12,13,14,15);

    const __bf16* bp = bp0 + kb;
    v8bf b00 = *(const v8bf*)(bp);                 v8bf b01 = *(const v8bf*)(bp + 16);
    v8bf b10 = *(const v8bf*)(bp + bstride);       v8bf b11 = *(const v8bf*)(bp + bstride + 16);
    v8bf b20 = *(const v8bf*)(bp + 2 * bstride);   v8bf b21 = *(const v8bf*)(bp + 2 * bstride + 16);
    v8bf b30 = *(const v8bf*)(bp + 3 * bstride);   v8bf b31 = *(const v8bf*)(bp + 3 * bstride + 16);
    v16bf b0 = __builtin_shufflevector(b00, b01, 0,1,2,3,4,5,6,7,8,9,10,11,12,13,14,15);
    v16bf b1 = __builtin_shufflevector(b10, b11, 0,1,2,3,4,5,6,7,8,9,10,11,12,13,14,15);
    v16bf b2 = __builtin_shufflevector(b20, b21, 0,1,2,3,4,5,6,7,8,9,10,11,12,13,14,15);
    v16bf b3 = __builtin_shufflevector(b30, b31, 0,1,2,3,4,5,6,7,8,9,10,11,12,13,14,15);

    acc0 = __builtin_amdgcn_wmma_f32_16x16x32_bf16(false, a, false, b0, (short)0, acc0, false, false);
    acc1 = __builtin_amdgcn_wmma_f32_16x16x32_bf16(false, a, false, b1, (short)0, acc1, false, false);
    acc2 = __builtin_amdgcn_wmma_f32_16x16x32_bf16(false, a, false, b2, (short)0, acc2, false, false);
    acc3 = __builtin_amdgcn_wmma_f32_16x16x32_bf16(false, a, false, b3, (short)0, acc3, false, false);
  }

  // C/D layout: VGPR r -> row tm + hi*8 + r, col = tile_n + l15
  float* cp = C + (size_t)(tm + (hi << 3)) * Nc + tn + l15;
#pragma unroll
  for (int r = 0; r < 8; ++r) {
    float* crow = cp + (size_t)r * Nc;
    crow[0]  = acc0[r];
    crow[16] = acc1[r];
    crow[32] = acc2[r];
    crow[48] = acc3[r];
  }
}

// ---------------- propagate: out = selfw*dinv^2*h (+ alpha*h0), then edge atomics ----------------
__global__ void k_prop_self(const float* __restrict__ h, const float* __restrict__ dinv,
                            const float* __restrict__ h0, float* __restrict__ out,
                            int n, int lgd, float selfw, float alpha) {
  size_t i = (size_t)blockIdx.x * blockDim.x + threadIdx.x;
  size_t total = (size_t)n << lgd;
  if (i >= total) return;
  float di = dinv[i >> lgd];
  float v = selfw * di * di * h[i];
  if (h0) v += alpha * h0[i];
  out[i] = v;
}

__global__ void k_prop_edges(const float* __restrict__ h, const int* __restrict__ src,
                             const int* __restrict__ dst, const float* __restrict__ enorm,
                             float* __restrict__ out, int e, int lgd4, float w) {
  size_t i = (size_t)blockIdx.x * blockDim.x + threadIdx.x;  // edge * (d/4) + chunk
  size_t total = (size_t)e << lgd4;
  if (i >= total) return;
  int ed = (int)(i >> lgd4);
  int c  = ((int)(i & (((size_t)1 << lgd4) - 1))) << 2;
  int d  = 4 << lgd4;
  int s = src[ed], t = dst[ed];
  float nw = w * enorm[ed];
  const float4 hv = *(const float4*)(h + (size_t)s * d + c);
  float* op = out + (size_t)t * d + c;
  atomicAdd(op + 0, nw * hv.x);
  atomicAdd(op + 1, nw * hv.y);
  atomicAdd(op + 2, nw * hv.z);
  atomicAdd(op + 3, nw * hv.w);
}

// ---------------- (h + bias) -> eval BatchNorm -> ReLU, in place ----------------
__global__ void k_bn_relu(float* __restrict__ h, const float* __restrict__ bias,
                          const float* __restrict__ gamma, const float* __restrict__ beta,
                          const float* __restrict__ mean, const float* __restrict__ var,
                          int n, int dmask) {
  size_t i = (size_t)blockIdx.x * blockDim.x + threadIdx.x;
  size_t total = (size_t)n * (size_t)(dmask + 1);
  if (i >= total) return;
  int c = (int)(i & (unsigned)dmask);
  float v = h[i] + bias[c];
  v = (v - mean[c]) * rsqrtf(var[c] + BN_EPS) * gamma[c] + beta[c];
  h[i] = fmaxf(v, 0.0f);
}

__global__ void k_add(const float* __restrict__ a, const float* __restrict__ b,
                      float* __restrict__ o, size_t cnt) {
  size_t i = (size_t)blockIdx.x * blockDim.x + threadIdx.x;
  if (i < cnt) o[i] = a[i] + b[i];
}

// ---------------- decoder: adj[e] = sigmoid(dot(z[row], z[col])), wave per edge ----------------
__global__ __launch_bounds__(256) void k_adj(const float* __restrict__ z,
                                             const int* __restrict__ row,
                                             const int* __restrict__ col,
                                             float* __restrict__ adj, int e) {
  int gw   = (int)((blockIdx.x * blockDim.x + threadIdx.x) >> 5);
  int lane = (int)(threadIdx.x & 31);
  if (gw >= e) return;
  int r = row[gw], c = col[gw];
  const float4 a = *(const float4*)(z + (size_t)r * LAT + lane * 4);
  const float4 b = *(const float4*)(z + (size_t)c * LAT + lane * 4);
  float s = a.x * b.x + a.y * b.y + a.z * b.z + a.w * b.w;
#pragma unroll
  for (int off = 16; off > 0; off >>= 1) s += __shfl_xor(s, off, 32);
  if (lane == 0) adj[gw] = 1.0f / (1.0f + expf(-s));
}

static inline unsigned cdiv(size_t a, size_t b) { return (unsigned)((a + b - 1) / b); }
static inline size_t alignup(size_t x) { return (x + 255) & ~(size_t)255; }

extern "C" void kernel_launch(void* const* d_in, const int* in_sizes, int n_in,
                              void* d_out, int out_size, void* d_ws, size_t ws_size,
                              hipStream_t stream) {
  const float* x      = (const float*)d_in[0];
  const int*   ei     = (const int*)  d_in[1];
  const float* Wproj  = (const float*)d_in[2];
  const float* W1     = (const float*)d_in[3];
  const float* b1     = (const float*)d_in[4];
  const float* gamma1 = (const float*)d_in[5];
  const float* beta1  = (const float*)d_in[6];
  const float* mean1  = (const float*)d_in[7];
  const float* var1   = (const float*)d_in[8];
  const float* W2     = (const float*)d_in[9];
  const float* b2     = (const float*)d_in[10];
  const float* gamma2 = (const float*)d_in[11];
  const float* beta2  = (const float*)d_in[12];
  const float* mean2  = (const float*)d_in[13];
  const float* var2   = (const float*)d_in[14];

  const int n = in_sizes[0] / IN_DIM;   // 50000
  const int e = in_sizes[1] / 2;        // 800000
  const int* src = ei;
  const int* dst = ei + e;

  // workspace carve (byte offsets, 256B aligned)
  char* base = (char*)d_ws;
  size_t off = 0;
  float*  dinv  = (float*)(base + off); off = alignup(off + sizeof(float) * n);
  float*  enorm = (float*)(base + off); off = alignup(off + sizeof(float) * e);
  float*  z0    = (float*)(base + off); off = alignup(off + sizeof(float) * (size_t)n * LAT);
  float*  bufA  = (float*)(base + off); off = alignup(off + sizeof(float) * (size_t)n * HID);
  float*  bufB  = (float*)(base + off); off = alignup(off + sizeof(float) * (size_t)n * HID);
  float*  bufC  = (float*)(base + off); off = alignup(off + sizeof(float) * (size_t)n * LAT);
  float*  bufD  = (float*)(base + off); off = alignup(off + sizeof(float) * (size_t)n * LAT);
  __bf16* xbf   = (__bf16*)(base + off); off = alignup(off + sizeof(__bf16) * (size_t)n * IN_DIM);
  __bf16* abf   = (__bf16*)(base + off); off = alignup(off + sizeof(__bf16) * (size_t)n * HID);
  __bf16* wt    = (__bf16*)(base + off); off = alignup(off + sizeof(__bf16) * (size_t)IN_DIM * LAT);

  float* adj  = (float*)d_out;
  float* zout = (float*)d_out + e;

  const int T = 256;

  // gcn_norm: deg (with self loop) -> dinv -> per-edge norm
  k_fill<<<cdiv(n, T), T, 0, stream>>>(dinv, 1.0f, (size_t)n);
  k_deg_edges<<<cdiv(e, T), T, 0, stream>>>(dst, dinv, e);
  k_dinv<<<cdiv(n, T), T, 0, stream>>>(dinv, n);
  k_enorm<<<cdiv(e, T), T, 0, stream>>>(src, dst, dinv, enorm, e);

  // z0 = x @ W_proj   (bf16 WMMA; convert inputs once)
  k_f32_to_bf16<<<cdiv((size_t)n * IN_DIM, T), T, 0, stream>>>(x, xbf, (size_t)n * IN_DIM);
  k_transpose_bf16<<<cdiv(IN_DIM * LAT, T), T, 0, stream>>>(Wproj, wt, IN_DIM, LAT);
  k_gemm_bf16<<<cdiv((size_t)(n / 16) * (LAT / 64), 8), T, 0, stream>>>(xbf, wt, z0, n, IN_DIM, LAT);

  // GCNConv1: h1 = relu(bn(P(z0 @ W1) + b1))
  k_f32_to_bf16<<<cdiv((size_t)n * LAT, T), T, 0, stream>>>(z0, abf, (size_t)n * LAT);
  k_transpose_bf16<<<cdiv(LAT * HID, T), T, 0, stream>>>(W1, wt, LAT, HID);
  k_gemm_bf16<<<cdiv((size_t)(n / 16) * (HID / 64), 8), T, 0, stream>>>(abf, wt, bufA, n, LAT, HID);
  k_prop_self<<<cdiv((size_t)n << 8, T), T, 0, stream>>>(bufA, dinv, nullptr, bufB, n, 8, 1.0f, 0.0f);
  k_prop_edges<<<cdiv((size_t)e << 6, T), T, 0, stream>>>(bufA, src, dst, enorm, bufB, e, 6, 1.0f);
  k_bn_relu<<<cdiv((size_t)n << 8, T), T, 0, stream>>>(bufB, b1, gamma1, beta1, mean1, var1, n, HID - 1);

  // GCNConv2: h2 = relu(bn(P(h1 @ W2) + b2))
  k_f32_to_bf16<<<cdiv((size_t)n * HID, T), T, 0, stream>>>(bufB, abf, (size_t)n * HID);
  k_transpose_bf16<<<cdiv(HID * LAT, T), T, 0, stream>>>(W2, wt, HID, LAT);
  k_gemm_bf16<<<cdiv((size_t)(n / 16) * (LAT / 64), 8), T, 0, stream>>>(abf, wt, bufC, n, HID, LAT);
  k_prop_self<<<cdiv((size_t)n << 7, T), T, 0, stream>>>(bufC, dinv, nullptr, bufD, n, 7, 1.0f, 0.0f);
  k_prop_edges<<<cdiv((size_t)e << 5, T), T, 0, stream>>>(bufC, src, dst, enorm, bufD, e, 5, 1.0f);
  k_bn_relu<<<cdiv((size_t)n << 7, T), T, 0, stream>>>(bufD, b2, gamma2, beta2, mean2, var2, n, LAT - 1);

  // APPNP: h_k = (1-alpha) * P(h_{k-1}) + alpha * h2 ; h2 in bufD, ping-pong bufC / bufA
  const float* prev = bufD;
  for (int k = 0; k < KPROP; ++k) {
    float* outb = (k & 1) ? bufA : bufC;
    k_prop_self<<<cdiv((size_t)n << 7, T), T, 0, stream>>>(
        prev, dinv, bufD, outb, n, 7, 1.0f - ALPHA, ALPHA);
    k_prop_edges<<<cdiv((size_t)e << 5, T), T, 0, stream>>>(
        prev, src, dst, enorm, outb, e, 5, 1.0f - ALPHA);
    prev = outb;
  }

  // z = z_refined + z0 ; adj = sigmoid(<z[row], z[col]>)
  k_add<<<cdiv((size_t)n * LAT, T), T, 0, stream>>>(prev, z0, zout, (size_t)n * LAT);
  k_adj<<<cdiv((size_t)e * 32, T), T, 0, stream>>>(zout, src, dst, adj, e);
}
